// MinecraftAgent_29051158790175
// MI455X (gfx1250) — matrix operand order repaired and checked by reading, not compiled, not run on hardware
//
#include <hip/hip_runtime.h>
#include <hip/hip_bf16.h>

// ---------------------------------------------------------------------------
// CDNA5 (gfx1250) CASL agent forward pass.
// All heavy GEMMs (conv1/2/3 as im2col-GEMM, linear, LSTM input transform)
// use v_wmma_f32_16x16x32_bf16 with 2x1 (conv) / 2x2 (linear, gates) register blocking; fragments load as
// 2x b128. LSTM recurrence runs in one workgroup with whh (transposed) in LDS.
// ---------------------------------------------------------------------------

typedef __bf16 bf16_t;
typedef __attribute__((ext_vector_type(16))) __bf16 v16bf;
typedef __attribute__((ext_vector_type(8)))  __bf16 v8bf;
typedef __attribute__((ext_vector_type(8)))  float  v8f;

union FragBF { v16bf v; struct { v8bf lo, hi; } p; bf16_t e[16]; };
union FragF  { v8f   v; float  e[8];  };
union PackBF { bf16_t h[2]; unsigned int u; };

__device__ __forceinline__ float sigmoidf(float v) { return 1.0f / (1.0f + expf(-v)); }
__device__ __forceinline__ float reluf(float v) { return v > 0.0f ? v : 0.0f; }

__device__ __forceinline__ unsigned int pack2(float a, float b) {
  PackBF t; t.h[0] = (bf16_t)a; t.h[1] = (bf16_t)b; return t.u;
}

__device__ __forceinline__ v8f wmma_bf16(v16bf a, v16bf b, v8f c) {
  return __builtin_amdgcn_wmma_f32_16x16x32_bf16(
      /*neg_a=*/false, a, /*neg_b=*/false, b,
      /*c_mod=*/(short)0, c, /*reuse_a=*/false, /*reuse_b=*/false);
}

// A fragment (16x32) from row-major [M x K]: lane holds row M=lane%16;
// K-halves {k0+kh..+7} and {k0+16+kh..+7}, kh = (lane>>4)*8  ->  2x b128.
__device__ __forceinline__ v16bf frag_a_rm(const bf16_t* A, int lda, int row0, int k0) {
  const int lane = threadIdx.x & 31;
  const bf16_t* base = A + (size_t)(row0 + (lane & 15)) * lda + k0 + ((lane >> 4) << 3);
  FragBF f;
  f.p.lo = *(const v8bf*)base;
  f.p.hi = *(const v8bf*)(base + 16);
  return f.v;
}

// B fragment (32x16) from N-major [N x K] storage (B[k][n] = W[n*ldw + k]):
// lane holds column N=lane%16; K-halves {k0+kh..+7} and {k0+16+kh..+7}, kh = (lane>>4)*8 -> 2x b128.
__device__ __forceinline__ v16bf frag_b_nm(const bf16_t* W, int ldw, int k0, int col0) {
  const int lane = threadIdx.x & 31;
  const bf16_t* base = W + (size_t)(col0 + (lane & 15)) * ldw + k0 + ((lane >> 4) << 3);
  FragBF f;
  f.p.lo = *(const v8bf*)base;
  f.p.hi = *(const v8bf*)(base + 16);
  return f.v;
}

// Epilogue: bias+relu, pack bf16 pairs into N-major LDS buffer ybuf[n][ldy].
__device__ __forceinline__ void epi_lds(bf16_t* ybuf, int ldy, int c0, int col0,
                                        const float* bias, v8f acc, int nmax) {
  const int lane = threadIdx.x & 31;
  const int n = col0 + (lane & 15);
  if (n < nmax) {
    FragF a; a.v = acc;
    const int cb = c0 + 8 * (lane >> 4);
#pragma unroll
    for (int r = 0; r < 8; r += 2) {
      int c = cb + r;
      *(unsigned int*)&ybuf[n * ldy + c] =
          pack2(reluf(a.e[r] + bias[c]), reluf(a.e[r + 1] + bias[c + 1]));
    }
  }
}

// ---------------------------------------------------------------------------
__global__ void cvt_bf16_kernel(const float* __restrict__ src, bf16_t* __restrict__ dst, int n2) {
  int i = blockIdx.x * 256 + threadIdx.x;   // one packed pair per thread
  if (i < n2) {
    *(unsigned int*)&dst[i * 2] = pack2(src[i * 2], src[i * 2 + 1]);
  }
}

// ---------------------------------------------------------------------------
// Per-frame conv tower: one workgroup per (frame, modality).
// LDS layout (dynamic, 162496 B):
//   [0      .. 28224)  x   f32  [84*84]
//   [28224  .. 53824)  y1  bf16 [400][32]   (spatial-major)
//   [53824  .. 152128) col bf16 (im2col, N-major; max 96*512)
//   [152128 .. 162496) y2  bf16 [81][64]    (spatial-major)
// ---------------------------------------------------------------------------
__global__ void __launch_bounds__(256) conv_pipeline_kernel(
    const float* __restrict__ x,       // [4096,2,84,84]
    const bf16_t* __restrict__ w1bf,   // [2][32][64]
    const bf16_t* __restrict__ w2bf,   // [2][64][512]
    const bf16_t* __restrict__ w3bf,   // [2][64][576]
    const float* __restrict__ b1v, const float* __restrict__ b1a,
    const float* __restrict__ b2v, const float* __restrict__ b2a,
    const float* __restrict__ b3v, const float* __restrict__ b3a,
    bf16_t* __restrict__ Fv,           // [4096,3136]
    bf16_t* __restrict__ Fa) {
  extern __shared__ char smem[];
  float*  xs  = (float*)smem;
  bf16_t* y1s = (bf16_t*)(smem + 28224);
  bf16_t* col = (bf16_t*)(smem + 53824);
  bf16_t* y2s = (bf16_t*)(smem + 152128);

  const int t = blockIdx.x;
  const int m = blockIdx.y;
  const int tid = threadIdx.x;
  const int wid = tid >> 5;
  const int lane = tid & 31;
  const int nlo = lane & 15;

  const float* b1 = m ? b1a : b1v;
  const float* b2 = m ? b2a : b2v;
  const float* b3 = m ? b3a : b3v;
  const bf16_t* A1 = w1bf + m * 2048;
  const bf16_t* A2 = w2bf + m * 32768;
  const bf16_t* A3 = w3bf + m * 36864;

  // stage 0: load input frame (float4)
  const float4* xin4 = (const float4*)(x + ((size_t)t * 2 + m) * 7056);
  float4* xs4 = (float4*)xs;
  for (int i = tid; i < 1764; i += 256) xs4[i] = xin4[i];
  __syncthreads();

  // stage 1: im2col conv1, N-major col[n][64], packed pair stores
  for (int i = tid; i < 12800; i += 256) {
    int base = i * 2;
    int n = base >> 6, k = base & 63;
    int oy = n / 20, ox = n - oy * 20;
    const float* xrow = xs + (ox << 2);
    float v0 = xrow[(oy * 4 + (k >> 3)) * 84 + (k & 7)];
    int k1 = k + 1;
    float v1 = xrow[(oy * 4 + (k1 >> 3)) * 84 + (k1 & 7)];
    *(unsigned int*)&col[base] = pack2(v0, v1);
  }
  __syncthreads();
  // conv1 WMMA: M=32 (one 2-tile block) x N=400 (25 tiles), K=64
  for (int p = wid; p < 25; p += 8) {
    const int col0 = p * 16;
    v8f acc0 = {}, acc1 = {};
    for (int k0 = 0; k0 < 64; k0 += 32) {
      v16bf b = frag_b_nm(col, 64, k0, col0);
      acc0 = wmma_bf16(frag_a_rm(A1, 64, 0, k0), b, acc0);
      acc1 = wmma_bf16(frag_a_rm(A1, 64, 16, k0), b, acc1);
    }
    epi_lds(y1s, 32, 0, col0, b1, acc0, 400);
    epi_lds(y1s, 32, 16, col0, b1, acc1, 400);
  }
  __syncthreads();

  // stage 2: im2col conv2 (4x4 stride 2 on [20,20,32]), N-major col[96][512]
  for (int i = tid; i < 24576; i += 256) {
    int base = i * 2;
    int n = base >> 9, k = base & 511;
    unsigned int pk = 0u;
    if (n < 81) {
      int ic = k >> 4, rem = k & 15;
      int oy = n / 9, ox = n - oy * 9;
      int s = (oy * 2 + (rem >> 2)) * 20 + (ox * 2 + (rem & 3));
      int k1 = k + 1, ic1 = k1 >> 4, rem1 = k1 & 15;
      int s1 = (oy * 2 + (rem1 >> 2)) * 20 + (ox * 2 + (rem1 & 3));
      PackBF pb; pb.h[0] = y1s[s * 32 + ic]; pb.h[1] = y1s[s1 * 32 + ic1];
      pk = pb.u;
    }
    *(unsigned int*)&col[base] = pk;
  }
  __syncthreads();
  // conv2 WMMA: M=64 (2 blocks of 32) x N=96 (6 tiles), K=512
  for (int p = wid; p < 12; p += 8) {
    const int row0 = (p & 1) * 32;
    const int col0 = (p >> 1) * 16;
    v8f acc0 = {}, acc1 = {};
    for (int k0 = 0; k0 < 512; k0 += 32) {
      v16bf b = frag_b_nm(col, 512, k0, col0);
      acc0 = wmma_bf16(frag_a_rm(A2, 512, row0, k0), b, acc0);
      acc1 = wmma_bf16(frag_a_rm(A2, 512, row0 + 16, k0), b, acc1);
    }
    epi_lds(y2s, 64, row0, col0, b2, acc0, 81);
    epi_lds(y2s, 64, row0 + 16, col0, b2, acc1, 81);
  }
  __syncthreads();

  // stage 3: im2col conv3 (3x3 stride 1 on [9,9,64]), N-major col[64][576]
  for (int i = tid; i < 18432; i += 256) {
    int base = i * 2;
    int n = base / 576, k = base - n * 576;
    unsigned int pk = 0u;
    if (n < 49) {
      int oy = n / 7, ox = n - oy * 7;
      int ic = k / 9, rem = k - ic * 9, ky = rem / 3, kx = rem - ky * 3;
      int k1 = k + 1, ic1 = k1 / 9, rem1 = k1 - ic1 * 9;
      int ky1 = rem1 / 3, kx1 = rem1 - ky1 * 3;
      PackBF pb;
      pb.h[0] = y2s[((oy + ky) * 9 + (ox + kx)) * 64 + ic];
      pb.h[1] = y2s[((oy + ky1) * 9 + (ox + kx1)) * 64 + ic1];
      pk = pb.u;
    }
    *(unsigned int*)&col[base] = pk;
  }
  __syncthreads();
  // conv3 WMMA: M=64 (2 blocks of 32) x N=64 (4 tiles), K=576
  bf16_t* Fo = (m ? Fa : Fv) + (size_t)t * 3136;
  for (int p = wid; p < 8; p += 8) {
    const int row0 = (p & 1) * 32;
    const int col0 = (p >> 1) * 16;
    v8f acc0 = {}, acc1 = {};
    for (int k0 = 0; k0 < 576; k0 += 32) {
      v16bf b = frag_b_nm(col, 576, k0, col0);
      acc0 = wmma_bf16(frag_a_rm(A3, 576, row0, k0), b, acc0);
      acc1 = wmma_bf16(frag_a_rm(A3, 576, row0 + 16, k0), b, acc1);
    }
    const int n = col0 + nlo;
    if (n < 49) {
      FragF a0; a0.v = acc0;
      FragF a1; a1.v = acc1;
      const int cb = row0 + 8 * (lane >> 4);
#pragma unroll
      for (int r = 0; r < 8; ++r) {
        int c0 = cb + r, c1 = cb + 16 + r;
        Fo[c0 * 49 + n] = (bf16_t)reluf(a0.e[r] + b3[c0]);
        Fo[c1 * 49 + n] = (bf16_t)reluf(a1.e[r] + b3[c1]);
      }
    }
  }
}

// ---------------------------------------------------------------------------
// features = relu(F[4096,3136] @ wl.T[3136,512] + bl)  (one modality)
// 2x2 register blocking: each wave owns a 32x32 output block.
// grid (128 Mblocks, 4), block 128 (4 waves -> 16 Nblocks)
// ---------------------------------------------------------------------------
__global__ void __launch_bounds__(128) linear_kernel(
    const bf16_t* __restrict__ F, const bf16_t* __restrict__ wlbf,
    const float* __restrict__ bl, float* __restrict__ out512) {
  const int row0 = blockIdx.x * 32;
  const int col0 = (blockIdx.y * 4 + (threadIdx.x >> 5)) * 32;
  const int lane = threadIdx.x & 31;
  const int half = lane >> 4, nlo = lane & 15;
  const bf16_t* aPF = F + (size_t)(row0 + lane) * 3136;
  const bf16_t* bPF = wlbf + (size_t)(col0 + lane) * 3136;
  v8f acc00 = {}, acc01 = {}, acc10 = {}, acc11 = {};
  for (int k0 = 0; k0 < 3136; k0 += 32) {
    if (k0 + 256 < 3136) {
      __builtin_prefetch(aPF + k0 + 256, 0, 0);
      __builtin_prefetch(bPF + k0 + 256, 0, 0);
    }
    v16bf a0 = frag_a_rm(F, 3136, row0, k0);
    v16bf a1 = frag_a_rm(F, 3136, row0 + 16, k0);
    v16bf b0 = frag_b_nm(wlbf, 3136, k0, col0);
    v16bf b1 = frag_b_nm(wlbf, 3136, k0, col0 + 16);
    acc00 = wmma_bf16(a0, b0, acc00);
    acc01 = wmma_bf16(a0, b1, acc01);
    acc10 = wmma_bf16(a1, b0, acc10);
    acc11 = wmma_bf16(a1, b1, acc11);
  }
  FragF q[4];
  q[0].v = acc00; q[1].v = acc01; q[2].v = acc10; q[3].v = acc11;
#pragma unroll
  for (int z = 0; z < 4; ++z) {
    const int rB = row0 + (z >> 1) * 16 + 8 * half;
    const int cI = col0 + (z & 1) * 16 + nlo;
    const float bb = bl[cI];
#pragma unroll
    for (int r = 0; r < 8; ++r)
      out512[(size_t)(rB + r) * 512 + cI] = reluf(q[z].e[r] + bb);
  }
}

// ---------------------------------------------------------------------------
// CASL attention + fusion. One wave per timestep; lane j = attention unit j.
// ---------------------------------------------------------------------------
__global__ void __launch_bounds__(256) attention_kernel(
    const float* __restrict__ vf, const float* __restrict__ af,
    const float* __restrict__ h0,
    const float* __restrict__ wv, const float* __restrict__ bv,
    const float* __restrict__ wa, const float* __restrict__ ba,
    const float* __restrict__ wsA, const float* __restrict__ bsA,
    const float* __restrict__ wo, const float* __restrict__ bo,
    bf16_t* __restrict__ fusedbf) {
  const int t = blockIdx.x * 8 + (threadIdx.x >> 5);
  const int j = threadIdx.x & 31;
  float s = bv[j] + ba[j] + bsA[j];
  const float4* h04 = (const float4*)h0;
  const float4* ws4 = (const float4*)(wsA + j * 128);
  for (int k = 0; k < 32; ++k) {
    float4 hq = h04[k], wq = ws4[k];
    s += hq.x * wq.x + hq.y * wq.y + hq.z * wq.z + hq.w * wq.w;
  }
  const float* vrow = vf + (size_t)t * 512;
  const float* arow = af + (size_t)t * 512;
  const float4* v4 = (const float4*)vrow;
  const float4* a4 = (const float4*)arow;
  const float4* wv4 = (const float4*)(wv + j * 512);
  const float4* wa4 = (const float4*)(wa + j * 512);
  for (int k = 0; k < 128; ++k) {
    float4 xv = v4[k], yv = wv4[k], xa = a4[k], ya = wa4[k];
    s += xv.x * yv.x + xv.y * yv.y + xv.z * yv.z + xv.w * yv.w;
    s += xa.x * ya.x + xa.y * ya.y + xa.z * ya.z + xa.w * ya.w;
  }
  const float act = tanhf(s);
  float p0 = act * wo[j];
  float p1 = act * wo[32 + j];
#pragma unroll
  for (int off = 16; off; off >>= 1) {
    p0 += __shfl_xor(p0, off);
    p1 += __shfl_xor(p1, off);
  }
  const float l0 = p0 + bo[0], l1 = p1 + bo[1];
  const float mx = fmaxf(l0, l1);
  const float e0 = expf(l0 - mx), e1 = expf(l1 - mx);
  const float w0 = e0 / (e0 + e1), w1 = e1 / (e0 + e1);
  bf16_t* frow = fusedbf + (size_t)t * 512;
  for (int p = j; p < 256; p += 32) {
    int k = p * 2;
    *(unsigned int*)&frow[k] = pack2(w0 * vrow[k] + w1 * arow[k],
                                     w0 * vrow[k + 1] + w1 * arow[k + 1]);
  }
}

// ---------------------------------------------------------------------------
// xg = fused[4096,512] @ wih.T[512,512] + bih + bhh   (WMMA, 2x2 blocking)
// grid (128, 4), block 128
// ---------------------------------------------------------------------------
__global__ void __launch_bounds__(128) gates_x_kernel(
    const bf16_t* __restrict__ fusedbf, const bf16_t* __restrict__ wihbf,
    const float* __restrict__ bih, const float* __restrict__ bhh,
    float* __restrict__ xg) {
  const int row0 = blockIdx.x * 32;
  const int col0 = (blockIdx.y * 4 + (threadIdx.x >> 5)) * 32;
  const int lane = threadIdx.x & 31;
  const int half = lane >> 4, nlo = lane & 15;
  v8f acc00 = {}, acc01 = {}, acc10 = {}, acc11 = {};
  for (int k0 = 0; k0 < 512; k0 += 32) {
    v16bf a0 = frag_a_rm(fusedbf, 512, row0, k0);
    v16bf a1 = frag_a_rm(fusedbf, 512, row0 + 16, k0);
    v16bf b0 = frag_b_nm(wihbf, 512, k0, col0);
    v16bf b1 = frag_b_nm(wihbf, 512, k0, col0 + 16);
    acc00 = wmma_bf16(a0, b0, acc00);
    acc01 = wmma_bf16(a0, b1, acc01);
    acc10 = wmma_bf16(a1, b0, acc10);
    acc11 = wmma_bf16(a1, b1, acc11);
  }
  FragF q[4];
  q[0].v = acc00; q[1].v = acc01; q[2].v = acc10; q[3].v = acc11;
#pragma unroll
  for (int z = 0; z < 4; ++z) {
    const int rB = row0 + (z >> 1) * 16 + 8 * half;
    const int cI = col0 + (z & 1) * 16 + nlo;
    const float bb = bih[cI] + bhh[cI];
#pragma unroll
    for (int r = 0; r < 8; ++r)
      xg[(size_t)(rB + r) * 512 + cI] = q[z].e[r] + bb;
  }
}

// ---------------------------------------------------------------------------
// Sequential LSTM scan: single workgroup, whh transposed in LDS (256 KB).
// LDS: whh_t[128][512] | h[128] | c[128] | g[512]  = 265216 B
// ---------------------------------------------------------------------------
__global__ void __launch_bounds__(512) lstm_scan_kernel(
    const float* __restrict__ xg, const float* __restrict__ done,
    const float* __restrict__ h0, const float* __restrict__ c0,
    const float* __restrict__ whh, float* __restrict__ out) {
  extern __shared__ char smem[];
  float* whh_s = (float*)smem;               // [k*512 + j]
  float* h_s = (float*)(smem + 262144);
  float* c_s = h_s + 128;
  float* g_s = c_s + 128;
  const int tid = threadIdx.x;

  // transpose-load whh: float4 over k
  const float4* whh4 = (const float4*)whh;   // [512][32]
  for (int i = tid; i < 16384; i += 512) {
    int j = i >> 5, k4 = i & 31;
    float4 w = whh4[i];
    whh_s[(k4 * 4 + 0) * 512 + j] = w.x;
    whh_s[(k4 * 4 + 1) * 512 + j] = w.y;
    whh_s[(k4 * 4 + 2) * 512 + j] = w.z;
    whh_s[(k4 * 4 + 3) * 512 + j] = w.w;
  }
  if (tid < 128) { h_s[tid] = h0[tid]; c_s[tid] = c0[tid]; }
  __syncthreads();

  for (int t = 0; t < 4096; ++t) {
    if (tid < 128) {
      float mm = 1.0f - done[t];
      h_s[tid] *= mm;
      c_s[tid] *= mm;
    }
    __syncthreads();
    float s = xg[(size_t)t * 512 + tid];
#pragma unroll 8
    for (int k = 0; k < 128; ++k) s += whh_s[k * 512 + tid] * h_s[k];
    g_s[tid] = s;
    __syncthreads();
    if (tid < 128) {
      float ig = sigmoidf(g_s[tid]);
      float fg = sigmoidf(g_s[128 + tid]);
      float gg = tanhf(g_s[256 + tid]);
      float og = sigmoidf(g_s[384 + tid]);
      float cn = fg * c_s[tid] + ig * gg;
      float hn = og * tanhf(cn);
      c_s[tid] = cn;
      h_s[tid] = hn;
      out[(size_t)t * 128 + tid] = hn;
    }
    __syncthreads();
  }
  if (tid < 128) {
    out[524288 + tid] = h_s[tid];        // hT
    out[524288 + 128 + tid] = c_s[tid];  // cT
  }
}

// ---------------------------------------------------------------------------
extern "C" void kernel_launch(void* const* d_in, const int* in_sizes, int n_in,
                              void* d_out, int out_size, void* d_ws, size_t ws_size,
                              hipStream_t stream) {
  (void)in_sizes; (void)n_in; (void)out_size; (void)ws_size;
  // inputs (setup_inputs dict order, nested dicts in insertion order)
  const float* x    = (const float*)d_in[0];
  const float* done = (const float*)d_in[1];
  const float* h0   = (const float*)d_in[2];
  const float* c0   = (const float*)d_in[3];
  const float* vw1 = (const float*)d_in[4];  const float* vb1 = (const float*)d_in[5];
  const float* vw2 = (const float*)d_in[6];  const float* vb2 = (const float*)d_in[7];
  const float* vw3 = (const float*)d_in[8];  const float* vb3 = (const float*)d_in[9];
  const float* vwl = (const float*)d_in[10]; const float* vbl = (const float*)d_in[11];
  const float* aw1 = (const float*)d_in[12]; const float* ab1 = (const float*)d_in[13];
  const float* aw2 = (const float*)d_in[14]; const float* ab2 = (const float*)d_in[15];
  const float* aw3 = (const float*)d_in[16]; const float* ab3 = (const float*)d_in[17];
  const float* awl = (const float*)d_in[18]; const float* abl = (const float*)d_in[19];
  const float* wv = (const float*)d_in[20];  const float* bv = (const float*)d_in[21];
  const float* wa = (const float*)d_in[22];  const float* ba = (const float*)d_in[23];
  const float* wsA = (const float*)d_in[24]; const float* bsA = (const float*)d_in[25];
  const float* wo = (const float*)d_in[26];  const float* bo = (const float*)d_in[27];
  const float* wih = (const float*)d_in[28]; const float* whh = (const float*)d_in[29];
  const float* bih = (const float*)d_in[30]; const float* bhh = (const float*)d_in[31];

  // workspace carve-up (256 B aligned)
  char* ws = (char*)d_ws;
  size_t off = 0;
  auto carve = [&](size_t bytes) -> void* {
    void* p = ws + off;
    off += (bytes + 255) & ~(size_t)255;
    return p;
  };
  bf16_t* w1bf   = (bf16_t*)carve(2 * 2048 * sizeof(bf16_t));
  bf16_t* w2bf   = (bf16_t*)carve(2 * 32768 * sizeof(bf16_t));
  bf16_t* w3bf   = (bf16_t*)carve(2 * 36864 * sizeof(bf16_t));
  bf16_t* wlbf_v = (bf16_t*)carve((size_t)1605632 * sizeof(bf16_t));
  bf16_t* wlbf_a = (bf16_t*)carve((size_t)1605632 * sizeof(bf16_t));
  bf16_t* wihbf  = (bf16_t*)carve((size_t)262144 * sizeof(bf16_t));
  bf16_t* Fv     = (bf16_t*)carve((size_t)4096 * 3136 * sizeof(bf16_t));
  bf16_t* Fa     = (bf16_t*)carve((size_t)4096 * 3136 * sizeof(bf16_t));
  float*  vf     = (float*)carve((size_t)4096 * 512 * sizeof(float));
  float*  af     = (float*)carve((size_t)4096 * 512 * sizeof(float));
  bf16_t* fusedbf= (bf16_t*)carve((size_t)4096 * 512 * sizeof(bf16_t));
  float*  xg     = (float*)carve((size_t)4096 * 512 * sizeof(float));

  auto cvt = [&](const float* src, bf16_t* dst, int n) {
    int n2 = n / 2;
    cvt_bf16_kernel<<<(n2 + 255) / 256, 256, 0, stream>>>(src, dst, n2);
  };
  cvt(vw1, w1bf, 2048);            cvt(aw1, w1bf + 2048, 2048);
  cvt(vw2, w2bf, 32768);           cvt(aw2, w2bf + 32768, 32768);
  cvt(vw3, w3bf, 36864);           cvt(aw3, w3bf + 36864, 36864);
  cvt(vwl, wlbf_v, 1605632);       cvt(awl, wlbf_a, 1605632);
  cvt(wih, wihbf, 262144);

  conv_pipeline_kernel<<<dim3(4096, 2), 256, 162496, stream>>>(
      x, w1bf, w2bf, w3bf, vb1, ab1, vb2, ab2, vb3, ab3, Fv, Fa);

  linear_kernel<<<dim3(128, 4), 128, 0, stream>>>(Fv, wlbf_v, vbl, vf);
  linear_kernel<<<dim3(128, 4), 128, 0, stream>>>(Fa, wlbf_a, abl, af);

  attention_kernel<<<512, 256, 0, stream>>>(
      vf, af, h0, wv, bv, wa, ba, wsA, bsA, wo, bo, fusedbf);

  gates_x_kernel<<<dim3(128, 4), 128, 0, stream>>>(fusedbf, wihbf, bih, bhh, xg);

  lstm_scan_kernel<<<1, 512, 265216, stream>>>(xg, done, h0, c0, whh, (float*)d_out);
}
